// ModalCompletionModule_59124519797327
// MI455X (gfx1250) — matrix operand order repaired and checked by reading, not compiled, not run on hardware
//
#include <hip/hip_runtime.h>
#include <hip/hip_bf16.h>
#include <math.h>
#include <stdint.h>

// Problem constants (from reference): B,S,H,NH,T = 128,512,768,8,20 ; HD=96
#define Bb  128
#define Ss  512
#define Hh  768
#define NHh 8
#define Tt  20
#define HDd 96

typedef float    v8f  __attribute__((ext_vector_type(8)));
typedef _Float16 v16h __attribute__((ext_vector_type(16)));
typedef _Float16 v8h  __attribute__((ext_vector_type(8)));

// ---------------------------------------------------------------------------
// CDNA5 helpers
// ---------------------------------------------------------------------------
__device__ inline uint32_t lds_off_u32(const void* p) {
  return (uint32_t)(uintptr_t)p;  // LDS aperture: low 32 bits are the LDS offset
}

// GVS-mode async copy: 16 bytes global -> LDS (tracked with ASYNCcnt)
__device__ inline void async_copy_b128(uint32_t lds_byte_off, const void* gbase,
                                       uint32_t gbyte_off) {
  asm volatile("global_load_async_to_lds_b128 %0, %1, %2"
               :: "v"(lds_byte_off), "v"(gbyte_off), "s"(gbase)
               : "memory");
}
__device__ inline void wait_async() {
  asm volatile("s_wait_asynccnt 0" ::: "memory");
}

__device__ inline v8f wmma_f16(v16h a, v16h b, v8f c) {
  return __builtin_amdgcn_wmma_f32_16x16x32_f16(
      /*neg_a=*/false, a, /*neg_b=*/false, b,
      /*c_mod=*/(short)0, c, /*reuse_a=*/false, /*reuse_b=*/false);
}

__device__ inline v16h combine8(v8h lo, v8h hi) {
  v16h o;
#pragma unroll
  for (int e = 0; e < 8; ++e) { o[e] = lo[e]; o[e + 8] = hi[e]; }
  return o;
}

// A fragment (16x32 f16) from row-major f16: rows r = lane%16, half h = lane>>4
// element e<8 -> K = 8h+e ; e>=8 -> K = 16+8h+(e-8)   => two 16B vector loads
__device__ inline v16h frag_a_rm(const _Float16* t, int ld) {
  const int lane = threadIdx.x & 31;
  const int r = lane & 15, h = lane >> 4;
  v8h lo = *(const v8h*)(t + r * ld + 8 * h);
  v8h hi = *(const v8h*)(t + r * ld + 16 + 8 * h);
  return combine8(lo, hi);
}

// B fragment (32x16 f16) from row-major f16 [n][k]: n = lane%16,
// e<8 -> K = 8h+e ; e>=8 -> K = 16+8h+(e-8)   => two 16B vector loads
__device__ inline v16h frag_b_rm(const _Float16* t, int ld) {
  const int lane = threadIdx.x & 31;
  const int n = lane & 15, h = lane >> 4;
  v8h lo = *(const v8h*)(t + n * ld + 8 * h);
  v8h hi = *(const v8h*)(t + n * ld + 16 + 8 * h);
  return combine8(lo, hi);
}

// C/D fragment (16x16 f32): lane L: col n = L%16; VGPR j -> row = j + 8*(L>>4)
template <typename F>
__device__ inline void store_frag_c(v8f c, F f) {
  const int lane = threadIdx.x & 31;
  const int n = lane & 15, hh = lane >> 4;
#pragma unroll
  for (int j = 0; j < 8; ++j) f(j + 8 * hh, n, c[j]);
}

// ---------------------------------------------------------------------------
// Kernel: f32 -> f16 transcode (weights, done once per launch)
// ---------------------------------------------------------------------------
__global__ void f16cvt_kernel(const float* __restrict__ src, _Float16* __restrict__ dst, int n) {
  int i = (blockIdx.x * 256 + threadIdx.x) * 4;
  if (i + 3 < n) {
#pragma unroll
    for (int j = 0; j < 4; ++j) dst[i + j] = (_Float16)src[i + j];
  }
}

// ---------------------------------------------------------------------------
// Kernel: masked mean pooling over the two S/2 halves
// ---------------------------------------------------------------------------
__global__ void pool_kernel(const float* __restrict__ ehs, float* __restrict__ pooled) {
  const int b = blockIdx.x, tid = threadIdx.x;
  for (int h = tid; h < Hh; h += 256) {
    const float* base = ehs + (size_t)b * Ss * Hh + h;
    float st = 0.f, si = 0.f;
    for (int s = 0; s < Ss / 2; ++s)  st += base[(size_t)s * Hh];
    for (int s = Ss / 2; s < Ss; ++s) si += base[(size_t)s * Hh];
    pooled[(size_t)b * Hh + h]        = st * (2.f / Ss);
    pooled[(size_t)(Bb + b) * Hh + h] = si * (2.f / Ss);
  }
}

// ---------------------------------------------------------------------------
// Kernel: WMMA GEMM  C[M,N] = act((A[M,K]f32 @ W[N,K]f16^T + bias[N]) * scale)
// 64x64 tile / block, 128 threads (4 waves x 16 rows). A tile async-staged f32
// then converted once to an f16 LDS tile; W tile async-staged directly as f16.
// act: 0=none, 1=leaky_relu(0.01), 2=relu. M,N %64==0, K %32==0.
// ---------------------------------------------------------------------------
__global__ void gemm_kernel(const float* __restrict__ A, const _Float16* __restrict__ W,
                            const float* __restrict__ bias, float* __restrict__ C,
                            int M, int N, int K, float scale, int act) {
  __shared__ float    Af[64 * 32];
  __shared__ _Float16 Ah[64 * 32];
  __shared__ _Float16 Wh[64 * 32];
  const int m0 = blockIdx.x * 64;
  const int n0 = blockIdx.y * 64;
  const int tid = threadIdx.x;       // 128 threads
  const int wave = tid >> 5;

  v8f acc[4] = {};
  const uint32_t af_lds = lds_off_u32(Af);
  const uint32_t wh_lds = lds_off_u32(Wh);

  for (int kk = 0; kk < K; kk += 32) {
    __syncthreads();
    // A tile: 64x32 f32 = 512 b128 transfers
#pragma unroll
    for (int r = 0; r < 4; ++r) {
      int i = tid + 128 * r;               // 0..511
      int row = i >> 3, seg = i & 7;
      async_copy_b128(af_lds + (uint32_t)(row * 32 + seg * 4) * 4u, A,
                      (uint32_t)(((size_t)(m0 + row) * K + kk + seg * 4) * 4));
    }
    // W tile: 64x32 f16 = 256 b128 transfers (8 halves each)
#pragma unroll
    for (int r = 0; r < 2; ++r) {
      int i = tid + 128 * r;               // 0..255
      int row = i >> 2, seg = i & 3;
      async_copy_b128(wh_lds + (uint32_t)(row * 32 + seg * 8) * 2u, W,
                      (uint32_t)(((size_t)(n0 + row) * K + kk + seg * 8) * 2));
    }
    wait_async();
    __syncthreads();
    // convert A tile to f16 once (16 contiguous elements per thread)
    {
      const int base = tid * 16;
#pragma unroll
      for (int j = 0; j < 16; ++j) Ah[base + j] = (_Float16)Af[base + j];
    }
    __syncthreads();

    v16h af = frag_a_rm(Ah + wave * 16 * 32, 32);
#pragma unroll
    for (int t = 0; t < 4; ++t) {
      v16h bf = frag_b_rm(Wh + t * 16 * 32, 32);
      acc[t] = wmma_f16(af, bf, acc[t]);
    }
  }

#pragma unroll
  for (int t = 0; t < 4; ++t) {
    store_frag_c(acc[t], [&](int r, int n, float v) {
      int gn = n0 + t * 16 + n;
      int gm = m0 + wave * 16 + r;
      float x = (v + bias[gn]) * scale;
      if (act == 1)      x = x > 0.f ? x : 0.01f * x;
      else if (act == 2) x = fmaxf(x, 0.f);
      C[(size_t)gm * N + gn] = x;
    });
  }
}

// ---------------------------------------------------------------------------
// Kernel: LayerNorm over last dim (one block per row), optional ReLU
// ---------------------------------------------------------------------------
__global__ void ln_kernel(const float* __restrict__ x, const float* __restrict__ g,
                          const float* __restrict__ be, int cols, int relu,
                          float* __restrict__ out) {
  __shared__ float red[256];
  const int row = blockIdx.x, tid = threadIdx.x;
  const float* xr = x + (size_t)row * cols;

  float s = 0.f;
  for (int c = tid; c < cols; c += 256) s += xr[c];
  red[tid] = s; __syncthreads();
  for (int off = 128; off > 0; off >>= 1) { if (tid < off) red[tid] += red[tid + off]; __syncthreads(); }
  const float mean = red[0] / (float)cols;
  __syncthreads();

  float v = 0.f;
  for (int c = tid; c < cols; c += 256) { float d = xr[c] - mean; v += d * d; }
  red[tid] = v; __syncthreads();
  for (int off = 128; off > 0; off >>= 1) { if (tid < off) red[tid] += red[tid + off]; __syncthreads(); }
  const float inv = rsqrtf(red[0] / (float)cols + 1e-5f);

  for (int c = tid; c < cols; c += 256) {
    float y = (xr[c] - mean) * inv * g[c] + be[c];
    if (relu) y = fmaxf(y, 0.f);
    out[(size_t)row * cols + c] = y;
  }
}

// ---------------------------------------------------------------------------
// Kernel: query = alpha*image_seed + (1-alpha)*text_seed + sinusoid PE
// ---------------------------------------------------------------------------
__global__ void blendq_kernel(const float* __restrict__ seeds, const float* __restrict__ rel,
                              float* __restrict__ query) {
  const int b = blockIdx.x, t = blockIdx.y, tid = threadIdx.x;
  const float a = rel[b];
  for (int h = tid; h < Hh; h += 256) {
    float ts = seeds[(size_t)b * Hh + h];
    float is = seeds[(size_t)(Bb + b) * Hh + h];
    float ih = (float)(h & ~1);
    float ang = (float)t * __powf(10000.f, -ih / (float)Hh);
    float pe = (h & 1) ? __cosf(ang) : __sinf(ang);
    query[((size_t)b * Tt + t) * Hh + h] = a * is + (1.f - a) * ts + pe;
  }
}

// ---------------------------------------------------------------------------
// Kernel: fused attention per (batch, modality). 256 threads = 8 waves = 8 heads.
// Chunk-projects K then V from ehs (keys = the modality's S/2 half) with f16
// WMMA, scores in LDS f16, softmax in LDS, P@V accumulated with WMMA. qh is
// pre-scaled by 1/sqrt(HD) at the Q-projection GEMM. Weights arrive as f16.
// Dynamic LDS layout (bytes), total 253952:
//   [0,49152)         chunk  : 16 x 768 f32   (async-to-LDS staging)
//   [49152,73728)     chunkh : 16 x 768 f16   (converted once per chunk)
//   [73728,122880)    qhp    : 32 x 768 f16   (query heads, rows 20..31 zero)
//   [122880,204800)   sc     : 8 x 20 x 256 f16 (scores -> softmax probs)
//   [204800,253952)   kv     : 8 x 3072 f16   (per-wave K/V scratch)
// ---------------------------------------------------------------------------
__global__ void attention_kernel(const float* __restrict__ ehs,
                                 const _Float16* __restrict__ taWin, const float* __restrict__ taBin,
                                 const _Float16* __restrict__ iaWin, const float* __restrict__ iaBin,
                                 const float* __restrict__ qh_text, const float* __restrict__ qh_img,
                                 float* __restrict__ ctx_text, float* __restrict__ ctx_img) {
  extern __shared__ char smem[];
  float*    chunk  = (float*)smem;                   // 16*768 f32
  _Float16* chunkh = (_Float16*)(smem + 49152);      // 16*768 f16
  _Float16* qhp    = (_Float16*)(smem + 73728);      // 32*768 f16
  _Float16* sc     = (_Float16*)(smem + 122880);     // 8*20*256 f16
  _Float16* kv     = (_Float16*)(smem + 204800);     // 8*3072 f16

  const int b   = blockIdx.x;
  const int m   = blockIdx.y;          // 0=text (keys 0..255), 1=image (256..511)
  const int tid = threadIdx.x;         // 256
  const int w   = tid >> 5;            // wave index == head index

  const _Float16* Win = m ? iaWin : taWin;
  const float*    Bin = m ? iaBin : taBin;
  const float*    qh  = m ? qh_img : qh_text;
  float*          ctx = m ? ctx_img : ctx_text;
  const _Float16* Wk = Win + (size_t)Hh * Hh;
  const _Float16* Wv = Win + (size_t)2 * Hh * Hh;
  const float*    bk = Bin + Hh;
  const float*    bv = Bin + 2 * Hh;

  // Stage padded query heads (f16)
  for (int i = tid; i < 32 * Hh; i += 256) {
    int t = i / Hh, h = i % Hh;
    float v = (t < Tt) ? qh[((size_t)b * Tt + t) * Hh + h] : 0.f;
    qhp[i] = (_Float16)v;
  }

  const uint32_t chunk_lds = lds_off_u32(chunk);
  const uint32_t key0 = (uint32_t)(((size_t)b * Ss + (size_t)m * 256) * Hh * 4);
  _Float16* myh = kv + (size_t)w * 3072;   // phase A: [key16][dim96]; phase C: [dim96][key32]

  // ---------------- Phase A: raw scores over 256 keys, 16 keys per chunk ----
  for (int c = 0; c < 16; ++c) {
    __syncthreads();
#pragma unroll
    for (int r = 0; r < 12; ++r) {           // 16x768 f32 = 3072 b128 xfers
      int i = tid + 256 * r;
      async_copy_b128(chunk_lds + (uint32_t)i * 16u, ehs,
                      key0 + (uint32_t)(c * 16 * Hh * 4) + (uint32_t)i * 16u);
    }
    wait_async();
    __syncthreads();
    // convert chunk to f16 once (3 x 16 contiguous elements per thread)
#pragma unroll
    for (int g = 0; g < 3; ++g) {
      int base = (tid + 256 * g) * 16;
#pragma unroll
      for (int j = 0; j < 16; ++j) chunkh[base + j] = (_Float16)chunk[base + j];
    }
    __syncthreads();

    // Kc = chunk @ Wk^T + bk  (N slice = this head's 96 dims), store [key][dim] f16
    v8f kc[6] = {};
    for (int kk = 0; kk < Hh; kk += 32) {
      v16h af = frag_a_rm(chunkh + kk, Hh);
#pragma unroll
      for (int t6 = 0; t6 < 6; ++t6) {
        v16h bf = frag_b_rm(Wk + (size_t)(w * 96 + t6 * 16) * Hh + kk, Hh);
        kc[t6] = wmma_f16(af, bf, kc[t6]);
      }
    }
#pragma unroll
    for (int t6 = 0; t6 < 6; ++t6)
      store_frag_c(kc[t6], [&](int r, int n, float v) {
        myh[r * 96 + t6 * 16 + n] = (_Float16)(v + bk[w * 96 + t6 * 16 + n]); });

    // scores = qh_head @ Kc^T (qh already scaled by 1/sqrt(HD))
    v8f sf[2] = {};
    for (int kk = 0; kk < 96; kk += 32) {
      v16h bf = frag_b_rm(myh + kk, 96);                         // [n=key][k=dim]
#pragma unroll
      for (int mt = 0; mt < 2; ++mt) {
        v16h af = frag_a_rm(qhp + (size_t)(mt * 16) * Hh + w * 96 + kk, Hh);
        sf[mt] = wmma_f16(af, bf, sf[mt]);
      }
    }
#pragma unroll
    for (int mt = 0; mt < 2; ++mt)
      store_frag_c(sf[mt], [&](int r, int n, float v) {
        int t = mt * 16 + r;
        if (t < Tt) sc[((size_t)w * Tt + t) * 256 + c * 16 + n] = (_Float16)v;
      });
  }
  __syncthreads();

  // ---------------- Phase B: softmax over 256 keys, 160 rows --------------
  if (tid < NHh * Tt) {
    _Float16* row = sc + (size_t)tid * 256;
    float mx = -1e30f;
    for (int i = 0; i < 256; ++i) mx = fmaxf(mx, (float)row[i]);
    float sum = 0.f;
    for (int i = 0; i < 256; ++i) sum += __expf((float)row[i] - mx);
    float inv = 1.f / sum;
    for (int i = 0; i < 256; ++i) row[i] = (_Float16)(__expf((float)row[i] - mx) * inv);
  }

  // ---------------- Phase C: out = P @ V, 32 keys per K-step ---------------
  v8f of[2][6] = {};
  for (int cc = 0; cc < 8; ++cc) {
    for (int sub = 0; sub < 2; ++sub) {
      __syncthreads();
      int kb = cc * 32 + sub * 16;
#pragma unroll
      for (int r = 0; r < 12; ++r) {
        int i = tid + 256 * r;
        async_copy_b128(chunk_lds + (uint32_t)i * 16u, ehs,
                        key0 + (uint32_t)(kb * Hh * 4) + (uint32_t)i * 16u);
      }
      wait_async();
      __syncthreads();
#pragma unroll
      for (int g = 0; g < 3; ++g) {
        int base = (tid + 256 * g) * 16;
#pragma unroll
        for (int j = 0; j < 16; ++j) chunkh[base + j] = (_Float16)chunk[base + j];
      }
      __syncthreads();

      v8f vc[6] = {};
      for (int kk = 0; kk < Hh; kk += 32) {
        v16h af = frag_a_rm(chunkh + kk, Hh);
#pragma unroll
        for (int t6 = 0; t6 < 6; ++t6) {
          v16h bf = frag_b_rm(Wv + (size_t)(w * 96 + t6 * 16) * Hh + kk, Hh);
          vc[t6] = wmma_f16(af, bf, vc[t6]);
        }
      }
      // store Vc transposed: [dim][key32] f16
#pragma unroll
      for (int t6 = 0; t6 < 6; ++t6)
        store_frag_c(vc[t6], [&](int r, int n, float v) {
          myh[(t6 * 16 + n) * 32 + sub * 16 + r] =
              (_Float16)(v + bv[w * 96 + t6 * 16 + n]); });
    }

    // of += P(20x32 keys) @ Vc(32x96)   (pad rows 20..31 produce discarded rows)
    v16h af2[2];
#pragma unroll
    for (int mt = 0; mt < 2; ++mt)
      af2[mt] = frag_a_rm(sc + ((size_t)w * Tt + mt * 16) * 256 + cc * 32, 256);
#pragma unroll
    for (int t6 = 0; t6 < 6; ++t6) {
      v16h bf = frag_b_rm(myh + t6 * 16 * 32, 32);   // [n=dim][k=key]
#pragma unroll
      for (int mt = 0; mt < 2; ++mt)
        of[mt][t6] = wmma_f16(af2[mt], bf, of[mt][t6]);
    }
  }

#pragma unroll
  for (int mt = 0; mt < 2; ++mt)
#pragma unroll
    for (int t6 = 0; t6 < 6; ++t6)
      store_frag_c(of[mt][t6], [&](int r, int n, float v) {
        int t = mt * 16 + r;
        if (t < Tt) ctx[((size_t)b * Tt + t) * Hh + w * 96 + t6 * 16 + n] = v;
      });
}

// ---------------------------------------------------------------------------
// Small elementwise kernels
// ---------------------------------------------------------------------------
__global__ void blend_cap_kernel(const float* __restrict__ capt, const float* __restrict__ capi,
                                 const float* __restrict__ rel, float* __restrict__ out) {
  const int row = blockIdx.x;       // b*T + t
  const float be = rel[row / Tt];
  for (int h = threadIdx.x; h < Hh; h += 256) {
    size_t idx = (size_t)row * Hh + h;
    out[idx] = be * capi[idx] + (1.f - be) * capt[idx];
  }
}

__global__ void mean_kernel(const float* __restrict__ cap, float* __restrict__ m) {
  const int b = blockIdx.x;
  for (int h = threadIdx.x; h < Hh; h += 256) {
    float s = 0.f;
    for (int t = 0; t < Tt; ++t) s += cap[((size_t)b * Tt + t) * Hh + h];
    m[(size_t)b * Hh + h] = s / (float)Tt;
  }
}

__global__ void quality_kernel(const float* __restrict__ hq, const float* __restrict__ w2,
                               const float* __restrict__ b2, float* __restrict__ out) {
  const int b = threadIdx.x;
  if (b >= Bb) return;
  float s = b2[0];
  for (int c = 0; c < Hh / 2; ++c) s += hq[(size_t)b * (Hh / 2) + c] * w2[c];
  out[b] = 1.f / (1.f + __expf(-s));
}

// ---------------------------------------------------------------------------
// Host launch
// ---------------------------------------------------------------------------
extern "C" void kernel_launch(void* const* d_in, const int* in_sizes, int n_in,
                              void* d_out, int out_size, void* d_ws, size_t ws_size,
                              hipStream_t stream) {
  (void)in_sizes; (void)n_in; (void)out_size; (void)ws_size;

  const float* ehs   = (const float*)d_in[0];
  const float* rel   = (const float*)d_in[1];
  const float* tW1   = (const float*)d_in[3];
  const float* tb1   = (const float*)d_in[4];
  const float* tg    = (const float*)d_in[5];
  const float* tbe   = (const float*)d_in[6];
  const float* tW2   = (const float*)d_in[7];
  const float* tb2   = (const float*)d_in[8];
  const float* iW1   = (const float*)d_in[9];
  const float* ib1   = (const float*)d_in[10];
  const float* ig    = (const float*)d_in[11];
  const float* ibe   = (const float*)d_in[12];
  const float* iW2   = (const float*)d_in[13];
  const float* ib2   = (const float*)d_in[14];
  const float* taWin = (const float*)d_in[15];
  const float* taBin = (const float*)d_in[16];
  const float* taWout= (const float*)d_in[17];
  const float* taBout= (const float*)d_in[18];
  const float* iaWin = (const float*)d_in[19];
  const float* iaBin = (const float*)d_in[20];
  const float* iaWout= (const float*)d_in[21];
  const float* iaBout= (const float*)d_in[22];
  const float* laW   = (const float*)d_in[23];
  const float* laB   = (const float*)d_in[24];
  const float* laG   = (const float*)d_in[25];
  const float* laBe  = (const float*)d_in[26];
  const float* qW1   = (const float*)d_in[27];
  const float* qB1   = (const float*)d_in[28];
  const float* qG    = (const float*)d_in[29];
  const float* qBe   = (const float*)d_in[30];
  const float* qW2   = (const float*)d_in[31];
  const float* qB2   = (const float*)d_in[32];

  const size_t BH  = (size_t)Bb * Hh;          // 98304
  const size_t BTH = (size_t)Bb * Tt * Hh;     // 1966080
  const size_t HH  = (size_t)Hh * Hh;          // 589824
  const size_t H3H = 3 * HH;                   // 1769472
  const size_t QH  = (size_t)(Hh / 2) * Hh;    // 294912

  float* ws      = (float*)d_ws;
  float* pooled  = ws;                 // 2*B*H  (text, image)
  float* h1      = pooled + 2 * BH;    // 2*B*H
  float* seeds   = h1 + 2 * BH;        // 2*B*H
  float* query   = seeds + 2 * BH;     // B*T*H
  float* qht     = query + BTH;        // B*T*H
  float* qhi     = qht + BTH;          // B*T*H
  float* ctxt    = qhi + BTH;          // B*T*H
  float* ctxi    = ctxt + BTH;         // B*T*H
  float* capt    = ctxi + BTH;         // B*T*H
  float* capi    = capt + BTH;         // B*T*H
  float* cappre  = capi + BTH;         // B*T*H
  float* caplin  = cappre + BTH;       // B*T*H
  float* mcap    = caplin + BTH;       // B*H
  float* hq      = mcap + BH;          // B*(H/2)

  // f16 weight arena after the f32 arrays
  _Float16* wh      = (_Float16*)(hq + BH / 2);
  _Float16* tW1h    = wh;              wh += HH;
  _Float16* iW1h    = wh;              wh += HH;
  _Float16* tW2h    = wh;              wh += HH;
  _Float16* iW2h    = wh;              wh += HH;
  _Float16* taWinh  = wh;              wh += H3H;
  _Float16* iaWinh  = wh;              wh += H3H;
  _Float16* taWouth = wh;              wh += HH;
  _Float16* iaWouth = wh;              wh += HH;
  _Float16* laWh    = wh;              wh += HH;
  _Float16* qW1h    = wh;              wh += QH;

  float* cap_out = (float*)d_out;                    // [B*T*H] cap
  float* q_out   = cap_out + BTH;                    // [B] quality

  const float qscale = 1.0f / sqrtf((float)HDd);

  // 0) transcode weights to f16 (consumed by every GEMM tile / attention WG)
  f16cvt_kernel<<<HH / 1024, 256, 0, stream>>>(tW1,   tW1h,   (int)HH);
  f16cvt_kernel<<<HH / 1024, 256, 0, stream>>>(iW1,   iW1h,   (int)HH);
  f16cvt_kernel<<<HH / 1024, 256, 0, stream>>>(tW2,   tW2h,   (int)HH);
  f16cvt_kernel<<<HH / 1024, 256, 0, stream>>>(iW2,   iW2h,   (int)HH);
  f16cvt_kernel<<<H3H / 1024, 256, 0, stream>>>(taWin, taWinh, (int)H3H);
  f16cvt_kernel<<<H3H / 1024, 256, 0, stream>>>(iaWin, iaWinh, (int)H3H);
  f16cvt_kernel<<<HH / 1024, 256, 0, stream>>>(taWout, taWouth, (int)HH);
  f16cvt_kernel<<<HH / 1024, 256, 0, stream>>>(iaWout, iaWouth, (int)HH);
  f16cvt_kernel<<<HH / 1024, 256, 0, stream>>>(laW,   laWh,   (int)HH);
  f16cvt_kernel<<<QH / 1024, 256, 0, stream>>>(qW1,   qW1h,   (int)QH);

  // 1) pooling
  pool_kernel<<<Bb, 256, 0, stream>>>(ehs, pooled);

  // 2) seed projections: h = leaky(pooled @ W1^T + b1); LN; seed = LN(h) @ W2^T + b2
  dim3 g1(Bb / 64, Hh / 64);
  gemm_kernel<<<g1, 128, 0, stream>>>(pooled,      tW1h, tb1, h1,      Bb, Hh, Hh, 1.f, 1);
  gemm_kernel<<<g1, 128, 0, stream>>>(pooled + BH, iW1h, ib1, h1 + BH, Bb, Hh, Hh, 1.f, 1);
  ln_kernel<<<Bb, 256, 0, stream>>>(h1,      tg, tbe, Hh, 0, h1);
  ln_kernel<<<Bb, 256, 0, stream>>>(h1 + BH, ig, ibe, Hh, 0, h1 + BH);
  gemm_kernel<<<g1, 128, 0, stream>>>(h1,      tW2h, tb2, seeds,      Bb, Hh, Hh, 1.f, 0);
  gemm_kernel<<<g1, 128, 0, stream>>>(h1 + BH, iW2h, ib2, seeds + BH, Bb, Hh, Hh, 1.f, 0);

  // 3) query = blend(seeds) + PE
  blendq_kernel<<<dim3(Bb, Tt), 256, 0, stream>>>(seeds, rel, query);

  // 4) Q projections (fold 1/sqrt(HD) into qh)
  dim3 g2((Bb * Tt) / 64, Hh / 64);
  gemm_kernel<<<g2, 128, 0, stream>>>(query, taWinh, taBin, qht, Bb * Tt, Hh, Hh, qscale, 0);
  gemm_kernel<<<g2, 128, 0, stream>>>(query, iaWinh, iaBin, qhi, Bb * Tt, Hh, Hh, qscale, 0);

  // 5) fused attention (K/V projection + softmax + P@V), 248KB dynamic LDS
  attention_kernel<<<dim3(Bb, 2), 256, 253952, stream>>>(
      ehs, taWinh, taBin, iaWinh, iaBin, qht, qhi, ctxt, ctxi);

  // 6) output projections
  gemm_kernel<<<g2, 128, 0, stream>>>(ctxt, taWouth, taBout, capt, Bb * Tt, Hh, Hh, 1.f, 0);
  gemm_kernel<<<g2, 128, 0, stream>>>(ctxi, iaWouth, iaBout, capi, Bb * Tt, Hh, Hh, 1.f, 0);

  // 7) blend, la projection, LN+ReLU -> cap output
  blend_cap_kernel<<<Bb * Tt, 256, 0, stream>>>(capt, capi, rel, cappre);
  gemm_kernel<<<g2, 128, 0, stream>>>(cappre, laWh, laB, caplin, Bb * Tt, Hh, Hh, 1.f, 0);
  ln_kernel<<<Bb * Tt, 256, 0, stream>>>(caplin, laG, laBe, Hh, 1, cap_out);

  // 8) quality head
  mean_kernel<<<Bb, 256, 0, stream>>>(cap_out, mcap);
  dim3 g3(Bb / 64, (Hh / 2) / 64);
  gemm_kernel<<<g3, 128, 0, stream>>>(mcap, qW1h, qB1, hq, Bb, Hh / 2, Hh, 1.f, 0);
  ln_kernel<<<Bb, 256, 0, stream>>>(hq, qG, qBe, Hh / 2, 1, hq);
  quality_kernel<<<1, 128, 0, stream>>>(hq, qW2, qB2, q_out);
}
